// UltraTransformer_50964081934992
// MI455X (gfx1250) — matrix operand order repaired and checked by reading, not compiled, not run on hardware
//
#include <hip/hip_runtime.h>

typedef __attribute__((ext_vector_type(16))) _Float16 v16h;
typedef __attribute__((ext_vector_type(8)))  _Float16 v8h;
typedef __attribute__((ext_vector_type(8)))  float    v8f;

#define N_HEADS 16
#define D_MODEL 1024
#define D_FF    4096
#define DK      64
#define MAX_REL 32
#define LN_EPS  1e-5f
#define B_BATCH 2
#define SEQ     2048
#define MROWS   (B_BATCH * SEQ)   // 4096 token rows

// ---------------------------------------------------------------------------
// WMMA wrapper: D = A(16x32 f16) * B(32x16 f16) + C(16x16 f32)
// ---------------------------------------------------------------------------
__device__ __forceinline__ v8f wmma_f16(v16h a, v16h b, v8f c) {
  return __builtin_amdgcn_wmma_f32_16x16x32_f16(
      /*neg_a=*/false, a, /*neg_b=*/false, b,
      /*c_mod=*/(short)0, c, /*reuse_a=*/false, /*reuse_b=*/false);
}

// A fragment (16x32, M x K), row-major source, leading dim ld (halfs).
// lane L: g=L>>4, r=L&15. elem e -> K=16g+e, contiguous in memory.
// Requires 16B-aligned rows (ld multiple of 8 halfs, base 16B aligned).
__device__ __forceinline__ v16h frag_a_load(const _Float16* base, int ld) {
  const int lane = threadIdx.x & 31;
  const int g = lane >> 4, r = lane & 15;
  const _Float16* p = base + r * ld + (g << 4);
  v8h lo = *(const v8h*)p;
  v8h hi = *(const v8h*)(p + 8);
  return __builtin_shufflevector(lo, hi, 0, 1, 2, 3, 4, 5, 6, 7,
                                 8, 9, 10, 11, 12, 13, 14, 15);
}

// B fragment (32x16, K x N) where logical B[k][n] = W[n][k] (W row-major NxK).
// lane L: g=L>>4, n=L&15. elem e -> K=16g+e, contiguous in memory.
__device__ __forceinline__ v16h frag_b_nk(const _Float16* base, int ld) {
  const int lane = threadIdx.x & 31;
  const int g = lane >> 4, n = lane & 15;
  const _Float16* p = base + n * ld + (g << 4);
  v8h lo = *(const v8h*)p;
  v8h hi = *(const v8h*)(p + 8);
  return __builtin_shufflevector(lo, hi, 0, 1, 2, 3, 4, 5, 6, 7,
                                 8, 9, 10, 11, 12, 13, 14, 15);
}

__device__ __forceinline__ float gelu_exact(float x) {
  return 0.5f * x * (1.0f + erff(x * 0.70710678118654752f));
}

// ---------------------------------------------------------------------------
// f32 -> f16 convert
// ---------------------------------------------------------------------------
__global__ __launch_bounds__(256) void cvt_f32_f16(const float* __restrict__ in,
                                                   _Float16* __restrict__ out, int n) {
  int i = blockIdx.x * 256 + threadIdx.x;
  if (i < n) out[i] = (_Float16)in[i];
}

// ---------------------------------------------------------------------------
// Tiled WMMA GEMM:  C[M,N] = A[M,K] * W[N,K]^T (+bias) (+gelu)
// Block: 128x128 tile, 256 threads (8 waves), each wave -> 32x64 outputs
// (2 A-frags x 4 B-frags = 8 accumulators, 8 WMMAs per 32-deep K step).
// EPI: 0=none, 1=+bias, 2=+bias+GELU.  OUTH: write f16 instead of f32.
// ---------------------------------------------------------------------------
template <int EPI, bool OUTH>
__global__ __launch_bounds__(256) void gemm_wmma(const _Float16* __restrict__ A,
                                                 const _Float16* __restrict__ W,
                                                 const float* __restrict__ bias,
                                                 void* __restrict__ out,
                                                 int M, int N, int K) {
  constexpr int BM = 128, BN = 128, BK = 32, LDT = BK + 8;
  __shared__ _Float16 As[BM][LDT];
  __shared__ _Float16 Ws[BN][LDT];
  const int tid = threadIdx.x;
  const int wid = tid >> 5;
  const int m0 = blockIdx.x * BM;
  const int n0 = blockIdx.y * BN;
  const int wm = (wid & 3) * 32;   // wave row offset in tile (0..96)
  const int wn = (wid >> 2) * 64;  // wave col offset in tile (0 or 64)
  v8f acc[2][4] = {};
  const int lrow = tid >> 1;         // 0..127
  const int lcol = (tid & 1) * 16;   // 0 or 16
  const _Float16* ag = A + (size_t)(m0 + lrow) * K + lcol;
  const _Float16* wg = W + (size_t)(n0 + lrow) * K + lcol;
  for (int k0 = 0; k0 < K; k0 += BK) {
    *(v8h*)&As[lrow][lcol]     = *(const v8h*)(ag + k0);
    *(v8h*)&As[lrow][lcol + 8] = *(const v8h*)(ag + k0 + 8);
    *(v8h*)&Ws[lrow][lcol]     = *(const v8h*)(wg + k0);
    *(v8h*)&Ws[lrow][lcol + 8] = *(const v8h*)(wg + k0 + 8);
    if (k0 + BK < K) {
      __builtin_prefetch(ag + k0 + BK, 0, 1);
      __builtin_prefetch(wg + k0 + BK, 0, 1);
    }
    __syncthreads();
    v16h a0 = frag_a_load(&As[wm][0], LDT);
    v16h a1 = frag_a_load(&As[wm + 16][0], LDT);
#pragma unroll
    for (int j = 0; j < 4; ++j) {
      v16h b = frag_b_nk(&Ws[wn + 16 * j][0], LDT);
      acc[0][j] = wmma_f16(a0, b, acc[0][j]);
      acc[1][j] = wmma_f16(a1, b, acc[1][j]);
    }
    __syncthreads();
  }
  const int lane = tid & 31, g = lane >> 4, n = lane & 15;
#pragma unroll
  for (int i = 0; i < 2; ++i) {
#pragma unroll
    for (int j = 0; j < 4; ++j) {
#pragma unroll
      for (int r = 0; r < 8; ++r) {
        int row = m0 + wm + 16 * i + r + 8 * g;
        int col = n0 + wn + 16 * j + n;
        float v = acc[i][j][r];
        if (EPI >= 1) v += bias[col];
        if (EPI == 2) v = gelu_exact(v);
        if (OUTH) ((_Float16*)out)[(size_t)row * N + col] = (_Float16)v;
        else      ((float*)out)[(size_t)row * N + col] = v;
      }
    }
  }
}

// ---------------------------------------------------------------------------
// RoPE on q,k + permute q,k,v from [B*S, H*dk] f32 to [B,H,S,dk] f16
// grid: (S, B*H), block: 64 (= dk)
// ---------------------------------------------------------------------------
__global__ __launch_bounds__(64) void rope_permute(const float* __restrict__ Qf,
                                                   const float* __restrict__ Kf,
                                                   const float* __restrict__ Vf,
                                                   _Float16* __restrict__ Qh,
                                                   _Float16* __restrict__ Kh,
                                                   _Float16* __restrict__ Vh) {
  const int s = blockIdx.x;
  const int bh = blockIdx.y;
  const int h = bh & (N_HEADS - 1);
  const int b = bh >> 4;
  const int d = threadIdx.x;
  const size_t src = ((size_t)b * SEQ + s) * D_MODEL + h * DK;
  const size_t dst = ((size_t)bh * SEQ + s) * DK + d;
  const int i = d & 31;  // frequency index
  float inv = __powf(10000.0f, -(float)i * (1.0f / 32.0f));
  float ang = (float)s * inv;
  float sn, cs;
  __sincosf(ang, &sn, &cs);
  float z1q = Qf[src + 2 * i], z2q = Qf[src + 2 * i + 1];
  float z1k = Kf[src + 2 * i], z2k = Kf[src + 2 * i + 1];
  float oq, ok;
  if (d < 32) { oq = z1q * cs - z2q * sn; ok = z1k * cs - z2k * sn; }
  else        { oq = z1q * sn + z2q * cs; ok = z1k * sn + z2k * cs; }
  Qh[dst] = (_Float16)oq;
  Kh[dst] = (_Float16)ok;
  Vh[dst] = (_Float16)Vf[src + d];
}

// ---------------------------------------------------------------------------
// Flash attention with relative-position bias.
// grid: (S/64, B*H), block: 128 (4 waves). Wave w owns q rows [q0+16w, +16).
// Q,K,V: f16 [B,H,S,dk]. Output: f16 [B*S, H*dk] (ready for wo projection).
// V is staged TRANSPOSED in LDS so the P*V B-fragment is a contiguous load.
// ---------------------------------------------------------------------------
__global__ __launch_bounds__(128) void flash_attn(const _Float16* __restrict__ Q,
                                                  const _Float16* __restrict__ Km,
                                                  const _Float16* __restrict__ Vm,
                                                  const float* __restrict__ rel_tab,
                                                  _Float16* __restrict__ Operm) {
  constexpr int LDK = DK + 8;   // 72 halfs = 144B rows (16B aligned)
  constexpr int LDV = 32 + 8;   // Vt rows: 40 halfs = 80B (16B aligned)
  constexpr int LDP = 32 + 8;
  __shared__ _Float16 Ks[32][LDK];     // [key][dk]
  __shared__ _Float16 Vt[DK][LDV];     // [dk][key]  (transposed)
  __shared__ _Float16 Ps[4][16][LDP];  // per-wave P tile
  __shared__ float relb[2 * MAX_REL - 1][N_HEADS];
  const int tid = threadIdx.x;
  const int bh = blockIdx.y;
  const int h = bh & (N_HEADS - 1);
  const int b = bh >> 4;
  const int q0 = blockIdx.x * 64;
  const int wid = tid >> 5;
  const int lane = tid & 31, g = lane >> 4, n = lane & 15;

  for (int i = tid; i < (2 * MAX_REL - 1) * N_HEADS; i += 128)
    relb[i >> 4][i & 15] = rel_tab[i];

  const _Float16* Qb = Q + ((size_t)bh * SEQ + q0 + wid * 16) * DK;
  v16h aq0 = frag_a_load(Qb, DK);        // dk 0..31
  v16h aq1 = frag_a_load(Qb + 32, DK);   // dk 32..63

  float m8[8], l8[8];
  v8f acc[4] = {};
#pragma unroll
  for (int r = 0; r < 8; ++r) { m8[r] = -1e30f; l8[r] = 0.0f; }
  const float scale = 0.125f;  // 1/sqrt(64)

  const int lr = tid >> 2;          // 0..31 (key within tile)
  const int lc = (tid & 3) * 16;    // 0,16,32,48 (dk chunk)

  for (int kb = 0; kb < SEQ; kb += 32) {
    const _Float16* kg = Km + ((size_t)bh * SEQ + kb + lr) * DK + lc;
    const _Float16* vg = Vm + ((size_t)bh * SEQ + kb + lr) * DK + lc;
    *(v8h*)&Ks[lr][lc]     = *(const v8h*)kg;
    *(v8h*)&Ks[lr][lc + 8] = *(const v8h*)(kg + 8);
    v8h v0 = *(const v8h*)vg;
    v8h v1 = *(const v8h*)(vg + 8);
#pragma unroll
    for (int e = 0; e < 8; ++e) {
      Vt[lc + e][lr] = v0[e];
      Vt[lc + 8 + e][lr] = v1[e];
    }
    __syncthreads();

    // scores for 32 keys: two 16x16 tiles, K-dim = dk = 64 -> 2 wmmas each
    v8f s0 = {}, s1 = {};
    s0 = wmma_f16(aq0, frag_b_nk(&Ks[0][0], LDK), s0);
    s0 = wmma_f16(aq1, frag_b_nk(&Ks[0][32], LDK), s0);
    s1 = wmma_f16(aq0, frag_b_nk(&Ks[16][0], LDK), s1);
    s1 = wmma_f16(aq1, frag_b_nk(&Ks[16][32], LDK), s1);

    // online softmax update (rows r+8g, key cols kb+n and kb+16+n)
#pragma unroll
    for (int r = 0; r < 8; ++r) {
      int qrow = q0 + wid * 16 + r + 8 * g;
      int rel0 = qrow - (kb + n);
      int rel1 = qrow - (kb + 16 + n);
      rel0 = rel0 < -(MAX_REL - 1) ? -(MAX_REL - 1) : (rel0 > MAX_REL - 1 ? MAX_REL - 1 : rel0);
      rel1 = rel1 < -(MAX_REL - 1) ? -(MAX_REL - 1) : (rel1 > MAX_REL - 1 ? MAX_REL - 1 : rel1);
      float v0s = s0[r] * scale + relb[rel0 + MAX_REL - 1][h];
      float v1s = s1[r] * scale + relb[rel1 + MAX_REL - 1][h];
      float mx = fmaxf(v0s, v1s);
#pragma unroll
      for (int msk = 8; msk >= 1; msk >>= 1) mx = fmaxf(mx, __shfl_xor(mx, msk, 32));
      float mnew = fmaxf(m8[r], mx);
      float p0 = __expf(v0s - mnew);
      float p1 = __expf(v1s - mnew);
      float sum = p0 + p1;
#pragma unroll
      for (int msk = 8; msk >= 1; msk >>= 1) sum += __shfl_xor(sum, msk, 32);
      float alpha = __expf(m8[r] - mnew);
      l8[r] = l8[r] * alpha + sum;
      m8[r] = mnew;
#pragma unroll
      for (int t = 0; t < 4; ++t) acc[t][r] *= alpha;
      Ps[wid][r + 8 * g][n] = (_Float16)p0;
      Ps[wid][r + 8 * g][16 + n] = (_Float16)p1;
    }

    // P(16x32) * V(32x64): 4 output 16x16 tiles along dk.
    // B[k][d] = V[key k][dk d] = Vt[d][k] -> contiguous frag_b_nk on Vt.
    v16h pa = frag_a_load(&Ps[wid][0][0], LDP);
#pragma unroll
    for (int t = 0; t < 4; ++t)
      acc[t] = wmma_f16(pa, frag_b_nk(&Vt[t * 16][0], LDV), acc[t]);
    __syncthreads();
  }

  // normalize and write out in [B*S, H*dk] layout
#pragma unroll
  for (int r = 0; r < 8; ++r) {
    int qrow = q0 + wid * 16 + r + 8 * g;
    float inv = 1.0f / l8[r];
    size_t row = (size_t)b * SEQ + qrow;
#pragma unroll
    for (int t = 0; t < 4; ++t)
      Operm[row * D_MODEL + h * DK + t * 16 + n] = (_Float16)(acc[t][r] * inv);
  }
}

// ---------------------------------------------------------------------------
// Fused residual + LayerNorm: out = LN(X+Y)*g + b; optional f16 copy.
// One block (256 threads) per row of D_MODEL=1024.
// ---------------------------------------------------------------------------
template <bool WH16>
__global__ __launch_bounds__(256) void residual_ln(const float* __restrict__ X,
                                                   const float* __restrict__ Y,
                                                   const float* __restrict__ gam,
                                                   const float* __restrict__ bet,
                                                   float* __restrict__ Out,
                                                   _Float16* __restrict__ Out16) {
  const int tid = threadIdx.x;
  const int row = blockIdx.x;
  const float* xr = X + (size_t)row * D_MODEL;
  const float* yr = Y + (size_t)row * D_MODEL;
  float v[4];
  float s = 0.0f, ss = 0.0f;
#pragma unroll
  for (int j = 0; j < 4; ++j) {
    int c = tid + j * 256;
    float t = xr[c] + yr[c];
    v[j] = t; s += t; ss += t * t;
  }
#pragma unroll
  for (int msk = 16; msk >= 1; msk >>= 1) {
    s += __shfl_xor(s, msk, 32);
    ss += __shfl_xor(ss, msk, 32);
  }
  __shared__ float ps[8], ps2[8], tot[2];
  if ((tid & 31) == 0) { ps[tid >> 5] = s; ps2[tid >> 5] = ss; }
  __syncthreads();
  if (tid == 0) {
    float a = 0.0f, b2 = 0.0f;
#pragma unroll
    for (int i = 0; i < 8; ++i) { a += ps[i]; b2 += ps2[i]; }
    tot[0] = a; tot[1] = b2;
  }
  __syncthreads();
  float mean = tot[0] * (1.0f / D_MODEL);
  float var = tot[1] * (1.0f / D_MODEL) - mean * mean;
  float rinv = rsqrtf(var + LN_EPS);
#pragma unroll
  for (int j = 0; j < 4; ++j) {
    int c = tid + j * 256;
    float o = (v[j] - mean) * rinv * gam[c] + bet[c];
    Out[(size_t)row * D_MODEL + c] = o;
    if (WH16) Out16[(size_t)row * D_MODEL + c] = (_Float16)o;
  }
}

// ---------------------------------------------------------------------------
// Host-side orchestration
// ---------------------------------------------------------------------------
extern "C" void kernel_launch(void* const* d_in, const int* in_sizes, int n_in,
                              void* d_out, int out_size, void* d_ws, size_t ws_size,
                              hipStream_t stream) {
  const float* x       = (const float*)d_in[0];
  const float* wq      = (const float*)d_in[1];
  const float* wk      = (const float*)d_in[2];
  const float* wv      = (const float*)d_in[3];
  const float* wo      = (const float*)d_in[4];
  const float* wo_b    = (const float*)d_in[5];
  const float* rel_tab = (const float*)d_in[6];
  const float* ln1_g   = (const float*)d_in[7];
  const float* ln1_b   = (const float*)d_in[8];
  const float* ln2_g   = (const float*)d_in[9];
  const float* ln2_b   = (const float*)d_in[10];
  const float* w1      = (const float*)d_in[11];
  const float* b1      = (const float*)d_in[12];
  const float* w2      = (const float*)d_in[13];
  const float* b2      = (const float*)d_in[14];

  char* ws = (char*)d_ws;
  size_t off = 0;
  auto alloc = [&](size_t bytes) -> char* {
    char* p = ws + off;
    off += (bytes + 255) & ~(size_t)255;
    return p;
  };
  const size_t MD = (size_t)MROWS * D_MODEL;   // 4M token-feature elements
  const size_t DD = (size_t)D_MODEL * D_MODEL; // 1M
  const size_t FD = (size_t)D_FF * D_MODEL;    // 4M

  _Float16* xb  = (_Float16*)alloc(MD * 2);
  _Float16* wqh = (_Float16*)alloc(DD * 2);
  _Float16* wkh = (_Float16*)alloc(DD * 2);
  _Float16* wvh = (_Float16*)alloc(DD * 2);
  _Float16* woh = (_Float16*)alloc(DD * 2);
  _Float16* w1h = (_Float16*)alloc(FD * 2);
  _Float16* w2h = (_Float16*)alloc(FD * 2);
  float* qf = (float*)alloc(MD * 4);
  float* kf = (float*)alloc(MD * 4);
  float* vf = (float*)alloc(MD * 4);
  _Float16* qh = (_Float16*)alloc(MD * 2);
  _Float16* kh = (_Float16*)alloc(MD * 2);
  _Float16* vh = (_Float16*)alloc(MD * 2);
  _Float16* operm = (_Float16*)alloc(MD * 2);
  float* oproj = (float*)alloc(MD * 4);
  float* hfull = (float*)alloc(MD * 4);
  _Float16* h16 = (_Float16*)alloc(MD * 2);
  // FFN buffers alias q/k/v fp32 region (free after rope_permute):
  _Float16* fbuf = (_Float16*)qf;  // M x F f16 = 32MB  (q+k region)
  float* f2 = vf;                  // M x D f32 = 16MB  (v region)

  auto cvt = [&](const float* src, _Float16* dst, size_t n) {
    cvt_f32_f16<<<(int)((n + 255) / 256), 256, 0, stream>>>(src, dst, (int)n);
  };
  cvt(x, xb, MD);
  cvt(wq, wqh, DD); cvt(wk, wkh, DD); cvt(wv, wvh, DD); cvt(wo, woh, DD);
  cvt(w1, w1h, FD); cvt(w2, w2h, FD);

  dim3 gDD(MROWS / 128, D_MODEL / 128);
  gemm_wmma<0, false><<<gDD, 256, 0, stream>>>(xb, wqh, nullptr, qf, MROWS, D_MODEL, D_MODEL);
  gemm_wmma<0, false><<<gDD, 256, 0, stream>>>(xb, wkh, nullptr, kf, MROWS, D_MODEL, D_MODEL);
  gemm_wmma<0, false><<<gDD, 256, 0, stream>>>(xb, wvh, nullptr, vf, MROWS, D_MODEL, D_MODEL);

  rope_permute<<<dim3(SEQ, B_BATCH * N_HEADS), 64, 0, stream>>>(qf, kf, vf, qh, kh, vh);

  flash_attn<<<dim3(SEQ / 64, B_BATCH * N_HEADS), 128, 0, stream>>>(qh, kh, vh, rel_tab, operm);

  gemm_wmma<1, false><<<gDD, 256, 0, stream>>>(operm, woh, wo_b, oproj, MROWS, D_MODEL, D_MODEL);

  residual_ln<true><<<MROWS, 256, 0, stream>>>(x, oproj, ln1_g, ln1_b, hfull, h16);

  gemm_wmma<2, true><<<dim3(MROWS / 128, D_FF / 128), 256, 0, stream>>>(
      h16, w1h, b1, fbuf, MROWS, D_FF, D_MODEL);
  gemm_wmma<1, false><<<gDD, 256, 0, stream>>>(fbuf, w2h, b2, f2, MROWS, D_MODEL, D_FF);

  residual_ln<false><<<MROWS, 256, 0, stream>>>(hfull, f2, ln2_g, ln2_b, (float*)d_out, nullptr);
}
